// EdgeSegmentSampler_57526791962714
// MI455X (gfx1250) — matrix operand order repaired and checked by reading, not compiled, not run on hardware
//
#include <hip/hip_runtime.h>

typedef __attribute__((ext_vector_type(16))) _Float16 v16h;
typedef __attribute__((ext_vector_type(8)))  _Float16 h8;
typedef __attribute__((ext_vector_type(2)))  _Float16 h2;
typedef __attribute__((ext_vector_type(8)))  float    v8f;

constexpr int NPTS    = 8192;
constexpr int BATCH   = 64;
constexpr int TILE    = 128;   // positions per workgroup
constexpr int STR     = 72;    // h1t row stride in halves (144 B, 16B aligned)
constexpr int HPOS    = 130;   // TILE + 2 halo (conv2 K=3, pad=1)
constexpr int PTSH    = 134;   // TILE + 6 halo (conv1 K=5 on top of conv2 halo)
constexpr int SEGL    = 50;
constexpr int NSAMP   = 20;
constexpr int VALID   = NPTS - SEGL;

static __device__ __forceinline__ v16h cat16(h8 a, h8 b)
{
    return __builtin_shufflevector(a, b, 0, 1, 2, 3, 4, 5, 6, 7,
                                         8, 9, 10, 11, 12, 13, 14, 15);
}

// ---------------------------------------------------------------------------
// Kernel 1: fused conv1(2->64,K5) -> ReLU -> conv2(64->32,K3 via WMMA) -> ReLU
//           -> conv3(32->1) -> sigmoid  + curvature  => combined (B,N) in ws
// GEMM K-dim is tap-major: k = tap*64 + ch, so each 32-wide k-step has a
// constant tap and 16 contiguous channels per lane -> 2x ds_load_b128 per frag.
// ---------------------------------------------------------------------------
__global__ __launch_bounds__(256)
void fused_importance_kernel(const float* __restrict__ points,
                             const float* __restrict__ w1,
                             const float* __restrict__ b1,
                             const float* __restrict__ w2,
                             const float* __restrict__ b2,
                             const float* __restrict__ w3,
                             const float* __restrict__ b3,
                             float* __restrict__ combined)
{
    __shared__ float                ptsld[PTSH * 2];
    __shared__ float                w1ld[64 * 10];
    __shared__ float                b1ld[64];
    __shared__ float                b2ld[32];
    __shared__ float                w3ld[32];
    __shared__ float                b3ld;
    __shared__ alignas(16) _Float16 w2ld[32 * 192];   // [m][k], k = tap*64 + ch
    __shared__ alignas(16) _Float16 h1t[HPOS * STR];  // [local pos][ch], pos0 = start-1

    const int tid   = threadIdx.x;
    const int b     = blockIdx.y;
    const int start = blockIdx.x * TILE;

    // ---------------- cooperative loads ----------------
    for (int i = tid; i < PTSH; i += 256) {
        int g = start - 3 + i;
        float x = 0.f, y = 0.f;
        if (g >= 0 && g < NPTS) {
            x = points[(b * NPTS + g) * 2 + 0];
            y = points[(b * NPTS + g) * 2 + 1];
        }
        ptsld[i * 2 + 0] = x;
        ptsld[i * 2 + 1] = y;
    }
    for (int i = tid; i < 640; i += 256) w1ld[i] = w1[i];
    // rearrange w2[m][ch][tap] -> w2ld[m][tap*64 + ch]
    for (int i = tid; i < 6144; i += 256) {
        int m   = i / 192;
        int r   = i - m * 192;
        int tap = r >> 6;            // r / 64
        int ch  = r & 63;            // r % 64
        w2ld[i] = (_Float16)w2[(m * 64 + ch) * 3 + tap];
    }
    if (tid < 64)            b1ld[tid]      = b1[tid];
    else if (tid < 96)       b2ld[tid - 64] = b2[tid - 64];
    else if (tid < 128)      w3ld[tid - 96] = w3[tid - 96];
    else if (tid == 128)     b3ld           = b3[0];
    __syncthreads();

    // ---------------- conv1 -> ReLU -> h1t (f16, pos-major), 2 ch / thread ----
    for (int i = tid; i < 32 * HPOS; i += 256) {
        int p  = i >> 5;             // local position 0..129
        int ch = (i & 31) * 2;       // even channel of the pair
        int g  = start - 1 + p;      // global position of this h1 entry
        float v0 = 0.f, v1 = 0.f;
        if (g >= 0 && g < NPTS) {    // conv2 zero-pad outside [0,N)
            float s0 = b1ld[ch], s1 = b1ld[ch + 1];
            const float* wx0 = &w1ld[ch * 10];
            const float* wx1 = wx0 + 10;
#pragma unroll
            for (int t = 0; t < 5; ++t) {   // zero-padded x via ptsld zero fill
                float px = ptsld[(p + t) * 2 + 0];
                float py = ptsld[(p + t) * 2 + 1];
                s0 += wx0[t] * px + wx0[5 + t] * py;
                s1 += wx1[t] * px + wx1[5 + t] * py;
            }
            v0 = s0 > 0.f ? s0 : 0.f;
            v1 = s1 > 0.f ? s1 : 0.f;
        }
        h2 pk;
        pk[0] = (_Float16)v0;
        pk[1] = (_Float16)v1;
        *(h2*)&h1t[p * STR + ch] = pk;      // one aligned 32-bit LDS store
    }
    __syncthreads();

    // ---------------- conv2 as WMMA GEMM: (32 x 192) x (192 x 16) ----------------
    const int wave = tid >> 5;
    const int lane = tid & 31;
    const int hh   = lane >> 4;      // half-wave: 0/1
    const int l16  = lane & 15;
    const int n0   = wave * 16;      // 8 waves cover TILE=128 positions

    v8f accLo, accHi;                // C/D: M = v + 8*hh (+16 for Hi), Npos = l16
#pragma unroll
    for (int v = 0; v < 8; ++v) {
        accLo[v] = b2ld[v + 8 * hh];
        accHi[v] = b2ld[16 + v + 8 * hh];
    }

#pragma unroll
    for (int ks = 0; ks < 6; ++ks) {        // 6 * 32 = 192 K-dim
        const int tap = ks >> 1;            // constant per unrolled step
        const int chb = (ks & 1) * 32 + 16 * hh;

        // B fragment: element j = B[k][n], k = ks*32 + 16*hh + j  ->
        //             ch = chb + j contiguous, pos = n0 + l16 + tap (halo=1)
        const _Float16* bp = &h1t[(n0 + l16 + tap) * STR + chb];
        v16h bf = cat16(*(const h8*)bp, *(const h8*)(bp + 8));

        // A fragments: element j = A[m][k], k = ks*32 + 16*hh + j
        const int kb = ks * 32 + 16 * hh;
        const _Float16* a0 = &w2ld[l16 * 192 + kb];
        const _Float16* a1 = a0 + 16 * 192;
        v16h aLo = cat16(*(const h8*)a0, *(const h8*)(a0 + 8));
        v16h aHi = cat16(*(const h8*)a1, *(const h8*)(a1 + 8));

        accLo = __builtin_amdgcn_wmma_f32_16x16x32_f16(false, aLo, false, bf,
                                                       (short)0, accLo, false, false);
        accHi = __builtin_amdgcn_wmma_f32_16x16x32_f16(false, aHi, false, bf,
                                                       (short)0, accHi, false, false);
    }

    // ---------------- ReLU -> conv3 -> sigmoid + curvature ----------------
    float part = 0.f;
#pragma unroll
    for (int v = 0; v < 8; ++v) {
        float lo = accLo[v]; lo = lo > 0.f ? lo : 0.f;
        float hi = accHi[v]; hi = hi > 0.f ? hi : 0.f;
        part += w3ld[v + 8 * hh] * lo + w3ld[16 + v + 8 * hh] * hi;
    }
    part += __shfl_xor(part, 16, 32);       // combine the two half-wave M-sets

    if (hh == 0) {
        int pos = start + n0 + l16;
        float imp = 1.f / (1.f + __expf(-(part + b3ld)));

        // curvature with circular rolls (matches jnp.roll)
        int im1 = (pos == 0) ? (NPTS - 1) : pos - 1;
        int ip1 = (pos == NPTS - 1) ? 0 : pos + 1;
        const float* P = points + (size_t)b * NPTS * 2;
        float cx = P[pos * 2], cy = P[pos * 2 + 1];
        float px = P[im1 * 2], py = P[im1 * 2 + 1];
        float nx = P[ip1 * 2], ny = P[ip1 * 2 + 1];
        float fx = nx - cx, fy = ny - cy;
        float bx = cx - px, by = cy - py;
        float curv = fabsf(fx * by - fy * bx);

        combined[(size_t)b * NPTS + pos] = imp + 0.3f * curv;
    }
}

// ---------------------------------------------------------------------------
// Kernel 2: sequential Gumbel-argmax sampling (== categorical over softmax),
// 20 steps, window decay on the LDS-resident score row. One WG per batch row.
// ---------------------------------------------------------------------------
__device__ __forceinline__ unsigned mix32(unsigned a)
{
    a ^= a >> 16; a *= 0x7feb352dU;
    a ^= a >> 15; a *= 0x846ca68bU;
    a ^= a >> 16;
    return a;
}

__global__ __launch_bounds__(256)
void sampler_kernel(const float* __restrict__ points,
                    const float* __restrict__ combined,
                    float* __restrict__ out)
{
    __shared__ float ci[NPTS];
    __shared__ float rv[256];
    __shared__ int   ri[256];
    __shared__ int   s_start;

    const int tid = threadIdx.x;
    const int b   = blockIdx.x;

    for (int i = tid; i < NPTS; i += 256)
        ci[i] = combined[(size_t)b * NPTS + i];
    __syncthreads();

    for (int s = 0; s < NSAMP; ++s) {
        // argmax_j ( ci[j] + Gumbel ) over j in [0, VALID)
        float best = -3.4e38f;
        int   bidx = 0;
        for (int j = tid; j < VALID; j += 256) {
            unsigned h = mix32((unsigned)j * 0x9E3779B9u ^
                               (unsigned)b * 0x85EBCA6Bu ^
                               (unsigned)(s + 1) * 0xC2B2AE3Du);
            float u = (float)(h >> 8) * (1.0f / 16777216.0f) + 1e-12f;
            float g = -__logf(-__logf(u));
            float v = ci[j] + g;
            if (v > best) { best = v; bidx = j; }
        }
        rv[tid] = best; ri[tid] = bidx;
        __syncthreads();
        for (int off = 128; off > 0; off >>= 1) {
            if (tid < off && rv[tid + off] > rv[tid]) {
                rv[tid] = rv[tid + off];
                ri[tid] = ri[tid + off];
            }
            __syncthreads();
        }
        if (tid == 0) s_start = ri[0];
        __syncthreads();

        int st = s_start;
        if (tid < SEGL) {
            int gp = st + tid;
            const float* P = points + ((size_t)b * NPTS + gp) * 2;
            size_t o = ((size_t)(s * BATCH + b) * SEGL + tid) * 2;
            out[o]     = P[0];
            out[o + 1] = P[1];
            ci[gp] *= 0.1f;                 // DECAY on sampled window
        }
        __syncthreads();
    }
}

// ---------------------------------------------------------------------------
extern "C" void kernel_launch(void* const* d_in, const int* in_sizes, int n_in,
                              void* d_out, int out_size, void* d_ws, size_t ws_size,
                              hipStream_t stream)
{
    (void)in_sizes; (void)n_in; (void)out_size; (void)ws_size;

    const float* points = (const float*)d_in[0];
    const float* w1     = (const float*)d_in[1];
    const float* b1     = (const float*)d_in[2];
    const float* w2     = (const float*)d_in[3];
    const float* b2     = (const float*)d_in[4];
    const float* w3     = (const float*)d_in[5];
    const float* b3     = (const float*)d_in[6];
    float*       out    = (float*)d_out;
    float*       comb   = (float*)d_ws;     // 64*8192 floats = 2 MiB scratch

    dim3 grid1(NPTS / TILE, BATCH);         // 64 x 64 workgroups, 256 thr (8 waves)
    fused_importance_kernel<<<grid1, 256, 0, stream>>>(points, w1, b1, w2, b2,
                                                       w3, b3, comb);
    sampler_kernel<<<BATCH, 256, 0, stream>>>(points, comb, out);
}
